// AttentionDecoder_27668179320935
// MI455X (gfx1250) — matrix operand order repaired and checked by reading, not compiled, not run on hardware
//
#include <hip/hip_runtime.h>

typedef __bf16 bf16;
typedef __attribute__((ext_vector_type(16))) __bf16 v16bf;
typedef __attribute__((ext_vector_type(8)))  float  v8f;

constexpr int cB = 256, cS = 128, cENC = 512, cDEC = 512, cEMB = 256, cV = 32000;

// ---------------- WMMA fragment helpers (wave32, contiguous k order: elem j of lane L is k = 16*(L>>4) + j) -------

// A: 16x32 bf16. lane L: row = L&15; holds K = kb..kb+15 (elems 0..15),
// kb = 16*(L>>4). Source fp32 row-major.
__device__ __forceinline__ v16bf frag_a(const float* A, int lda, int row0, int k0) {
  int lane = threadIdx.x & 31;
  const float* p = A + (size_t)(row0 + (lane & 15)) * lda + k0 + ((lane >> 4) << 4);
  v16bf f;
#pragma unroll
  for (int j = 0; j < 8; ++j) f[j] = (bf16)p[j];
#pragma unroll
  for (int j = 0; j < 8; ++j) f[j + 8] = (bf16)p[j + 8];
  return f;
}

// B: 32x16 bf16 from row-major KxN fp32. lane L: col = L&15, K base = 16*(L>>4).
__device__ __forceinline__ v16bf frag_b_rm(const float* Bm, int ldb, int k0, int col0) {
  int lane = threadIdx.x & 31;
  const float* p = Bm + (size_t)(k0 + ((lane >> 4) << 4)) * ldb + (col0 + (lane & 15));
  v16bf f;
#pragma unroll
  for (int j = 0; j < 16; ++j) f[j] = (bf16)p[(size_t)j * ldb];
  return f;
}

// B = W^T where W is row-major NxK fp32 (PyTorch Linear weight): B[k][n]=W[n][k].
// Per lane this is 16 contiguous floats of row `col`.
__device__ __forceinline__ v16bf frag_b_tr(const float* W, int ldw, int k0, int col0) {
  int lane = threadIdx.x & 31;
  const float* p = W + (size_t)(col0 + (lane & 15)) * ldw + k0 + ((lane >> 4) << 4);
  v16bf f;
#pragma unroll
  for (int j = 0; j < 16; ++j) f[j] = (bf16)p[j];
  return f;
}

__device__ __forceinline__ v8f wmma_bf16(v16bf a, v16bf b, v8f c) {
  return __builtin_amdgcn_wmma_f32_16x16x32_bf16(false, a, false, b, (short)0, c,
                                                 false, false);
}

// ---------------- Kernels ---------------------------------------------------

// emb[b][e] = emb_table[tok[b]][e]     grid=cB, block=cEMB
__global__ void k_gather_emb(const int* __restrict__ tok,
                             const float* __restrict__ table,
                             float* __restrict__ emb) {
  int b = blockIdx.x, t = threadIdx.x;
  emb[b * cEMB + t] = table[(size_t)tok[b] * cEMB + t];
}

// C[M,N] = X[M,K] @ Bm[K,N] + bias.   One wave per 16x16 tile.
__global__ __launch_bounds__(32)
void k_gemm_rm(const float* __restrict__ X, const float* __restrict__ Bm,
               const float* __restrict__ bias, float* __restrict__ C,
               int M, int N, int K) {
  (void)M;
  int ntn = N >> 4;
  int tn = blockIdx.x % ntn, tm = blockIdx.x / ntn;
  int row0 = tm << 4, col0 = tn << 4;
  int lane = threadIdx.x;
  v8f acc = {};
  for (int k = 0; k < K; k += 32)
    acc = wmma_bf16(frag_a(X, K, row0, k), frag_b_rm(Bm, N, k, col0), acc);
  int col = col0 + (lane & 15);
  float bv = bias ? bias[col] : 0.f;
  int rbase = row0 + ((lane >> 4) << 3);
#pragma unroll
  for (int r = 0; r < 8; ++r)
    C[(size_t)(rbase + r) * N + col] = acc[r] + bv;
}

// C[M,N] = X[M,K] @ W[N,K]^T + bias.  One wave handles MT row-tiles so each
// B-fragment load (weight stream) feeds MT wmma ops -> MT-way weight reuse.
// For MT=16 with M=256 the full weight matrix is streamed exactly once.
template <int MT>
__global__ __launch_bounds__(32)
void k_gemm_tr(const float* __restrict__ X, const float* __restrict__ W,
               const float* __restrict__ bias, float* __restrict__ C,
               int M, int N, int K) {
  (void)M;
  int ntn = N >> 4;
  int tn = blockIdx.x % ntn, tg = blockIdx.x / ntn;
  int col0 = tn << 4;
  int lane = threadIdx.x;
  v8f acc[MT];
#pragma unroll
  for (int i = 0; i < MT; ++i) acc[i] = (v8f){};
  for (int k = 0; k < K; k += 32) {
    v16bf bf = frag_b_tr(W, K, k, col0);
    if (k + 32 < K) {
      // prefetch this lane's next 64B chunk of the weight row (CDNA5
      // global_prefetch_b8); hides HBM latency behind the MT wmma ops below
      const float* np =
          W + (size_t)(col0 + (lane & 15)) * K + (k + 32) + ((lane >> 4) << 4);
      __builtin_prefetch(np, 0, 1);
    }
#pragma unroll
    for (int i = 0; i < MT; ++i) {
      v16bf af = frag_a(X, K, (tg * MT + i) << 4, k);
      acc[i] = wmma_bf16(af, bf, acc[i]);
    }
  }
  int col = col0 + (lane & 15);
  float bv = bias ? bias[col] : 0.f;
#pragma unroll
  for (int i = 0; i < MT; ++i) {
    int rbase = ((tg * MT + i) << 4) + ((lane >> 4) << 3);
#pragma unroll
    for (int r = 0; r < 8; ++r)
      C[(size_t)(rbase + r) * N + col] = acc[i][r] + bv;
  }
}

// Fused: energy = tanh(enc_flat @ Wenc + hW), scores[b][s] = energy . v
// Rows are flat (s*B+b); block = 1 wave handles 16 rows x all 512 cols.
__global__ __launch_bounds__(32)
void k_scores(const float* __restrict__ enc, const float* __restrict__ Wenc,
              const float* __restrict__ hW, const float* __restrict__ v,
              float* __restrict__ scores) {
  int row0 = blockIdx.x << 4;   // flat (s*cB + b), 16-aligned; cB%16==0 => same s
  int s  = row0 / cB;
  int b0 = row0 % cB;
  int lane = threadIdx.x;
  float sacc[8];
#pragma unroll
  for (int r = 0; r < 8; ++r) sacc[r] = 0.f;

  for (int col0 = 0; col0 < cDEC; col0 += 16) {
    v8f acc = {};
    for (int k = 0; k < cENC; k += 32)
      acc = wmma_bf16(frag_a(enc, cENC, row0, k), frag_b_rm(Wenc, cDEC, k, col0), acc);
    int col = col0 + (lane & 15);
    float vv = v[col];
    int mb = (lane >> 4) << 3;
#pragma unroll
    for (int r = 0; r < 8; ++r) {
      int bb = b0 + mb + r;
      float e = tanhf(acc[r] + hW[(size_t)bb * cDEC + col]);
      sacc[r] += vv * e;
    }
  }
  // reduce across the 16 lanes that share a row group (xor masks < 16)
#pragma unroll
  for (int r = 0; r < 8; ++r) {
    float t = sacc[r];
    t += __shfl_xor(t, 1, 32);
    t += __shfl_xor(t, 2, 32);
    t += __shfl_xor(t, 4, 32);
    t += __shfl_xor(t, 8, 32);
    sacc[r] = t;
  }
  if ((lane & 15) == 0) {
    int mb = (lane >> 4) << 3;
#pragma unroll
    for (int r = 0; r < 8; ++r)
      scores[(size_t)(b0 + mb + r) * cS + s] = sacc[r];
  }
}

// softmax over S per row b, then weighted[b][e] = sum_s a[s] * enc[s][b][e]
__global__ __launch_bounds__(256)
void k_softmax_weighted(const float* __restrict__ scores,
                        const float* __restrict__ enc,
                        float* __restrict__ weighted) {
  __shared__ float a[cS];
  __shared__ float red[256];
  int b = blockIdx.x, t = threadIdx.x;
  float sv = (t < cS) ? scores[b * cS + t] : -1e30f;
  red[t] = sv; __syncthreads();
  for (int off = 128; off > 0; off >>= 1) {
    if (t < off) red[t] = fmaxf(red[t], red[t + off]);
    __syncthreads();
  }
  float m = red[0]; __syncthreads();
  float e = (t < cS) ? __expf(sv - m) : 0.f;
  red[t] = e; __syncthreads();
  for (int off = 128; off > 0; off >>= 1) {
    if (t < off) red[t] += red[t + off];
    __syncthreads();
  }
  float inv = 1.f / red[0]; __syncthreads();
  if (t < cS) a[t] = e * inv;
  __syncthreads();
  for (int ei = t; ei < cENC; ei += 256) {
    float acc = 0.f;
    for (int s = 0; s < cS; ++s)
      acc += a[s] * enc[((size_t)s * cB + b) * cENC + ei];
    weighted[b * cENC + ei] = acc;
  }
}

// xin = [emb | weighted]  shape [B, EMB+ENC]
__global__ void k_build_xin(const float* __restrict__ emb,
                            const float* __restrict__ weighted,
                            float* __restrict__ xin) {
  int idx = blockIdx.x * blockDim.x + threadIdx.x;
  int b = idx / (cEMB + cENC), c = idx % (cEMB + cENC);
  xin[idx] = (c < cEMB) ? emb[b * cEMB + c] : weighted[b * cENC + (c - cEMB)];
}

// GRU combine: h_new = (1-z)*n + z*h ; write d_out tail and xcat[:, 0:DEC]
__global__ void k_gru_combine(const float* __restrict__ gi,
                              const float* __restrict__ gh,
                              const float* __restrict__ h,
                              float* __restrict__ out_h,
                              float* __restrict__ xcat) {
  int idx = blockIdx.x * blockDim.x + threadIdx.x;  // b*DEC + d
  int b = idx / cDEC, d = idx % cDEC;
  const float* gib = gi + (size_t)b * 3 * cDEC;
  const float* ghb = gh + (size_t)b * 3 * cDEC;
  float r  = 1.f / (1.f + __expf(-(gib[d] + ghb[d])));
  float z  = 1.f / (1.f + __expf(-(gib[cDEC + d] + ghb[cDEC + d])));
  float n  = tanhf(gib[2 * cDEC + d] + r * ghb[2 * cDEC + d]);
  float hn = (1.f - z) * n + z * h[idx];
  out_h[idx] = hn;
  xcat[(size_t)b * (cDEC + cENC + cEMB) + d] = hn;
}

// xcat[:, DEC:DEC+ENC] = weighted ; xcat[:, DEC+ENC:] = emb
__global__ void k_build_xcat_rest(const float* __restrict__ weighted,
                                  const float* __restrict__ emb,
                                  float* __restrict__ xcat) {
  int idx = blockIdx.x * blockDim.x + threadIdx.x;
  int b = idx / (cENC + cEMB), c = idx % (cENC + cEMB);
  float val = (c < cENC) ? weighted[b * cENC + c] : emb[b * cEMB + (c - cENC)];
  xcat[(size_t)b * (cDEC + cENC + cEMB) + cDEC + c] = val;
}

// ---------------- Launch -----------------------------------------------------

extern "C" void kernel_launch(void* const* d_in, const int* in_sizes, int n_in,
                              void* d_out, int out_size, void* d_ws, size_t ws_size,
                              hipStream_t stream) {
  (void)in_sizes; (void)n_in; (void)out_size; (void)ws_size;
  const int*   tok    = (const int*)d_in[0];
  const float* hidden = (const float*)d_in[1];   // [1,B,DEC] -> [B,DEC]
  const float* enc    = (const float*)d_in[2];   // [S,B,ENC] flat = (S*B, ENC)
  const float* table  = (const float*)d_in[3];
  const float* attn_W = (const float*)d_in[4];   // [ENC+DEC, DEC], h-rows first
  const float* attn_b = (const float*)d_in[5];
  const float* vvec   = (const float*)d_in[6];
  const float* W_ih   = (const float*)d_in[7];   // [3*DEC, EMB+ENC]
  const float* W_hh   = (const float*)d_in[8];   // [3*DEC, DEC]
  const float* b_ih   = (const float*)d_in[9];
  const float* b_hh   = (const float*)d_in[10];
  const float* fc_W   = (const float*)d_in[11];  // [V, DEC+ENC+EMB]
  const float* fc_b   = (const float*)d_in[12];

  float* out  = (float*)d_out;          // pred [B,V] then h_new [B,DEC]
  float* outh = out + (size_t)cB * cV;

  float* ws       = (float*)d_ws;
  float* emb      = ws;                                  // B*EMB
  float* hW       = emb + cB * cEMB;                     // B*DEC
  float* scores   = hW + cB * cDEC;                      // B*S
  float* weighted = scores + cB * cS;                    // B*ENC
  float* xin      = weighted + cB * cENC;                // B*(EMB+ENC)
  float* gi       = xin + cB * (cEMB + cENC);            // B*3*DEC
  float* gh       = gi + cB * 3 * cDEC;                  // B*3*DEC
  float* xcat     = gh + cB * 3 * cDEC;                  // B*(DEC+ENC+EMB)

  // 1) embedding gather
  k_gather_emb<<<dim3(cB), dim3(cEMB), 0, stream>>>(tok, table, emb);

  // 2) hW = h @ attn_W[0:DEC] + attn_b    (bias folded in once)
  k_gemm_rm<<<dim3((cB / 16) * (cDEC / 16)), dim3(32), 0, stream>>>(
      hidden, attn_W, attn_b, hW, cB, cDEC, cDEC);

  // 3) fused energy GEMM + tanh + (.v)  -> scores [B,S]
  k_scores<<<dim3((cS * cB) / 16), dim3(32), 0, stream>>>(
      enc, attn_W + (size_t)cDEC * cDEC, hW, vvec, scores);

  // 4) softmax + weighted context
  k_softmax_weighted<<<dim3(cB), dim3(256), 0, stream>>>(scores, enc, weighted);

  // 5) xin = [emb | weighted]
  k_build_xin<<<dim3(cB * (cEMB + cENC) / 256), dim3(256), 0, stream>>>(
      emb, weighted, xin);

  // 6) GRU gates: gi = xin @ W_ih^T + b_ih ; gh = h @ W_hh^T + b_hh
  k_gemm_tr<8><<<dim3((cB / 128) * (3 * cDEC / 16)), dim3(32), 0, stream>>>(
      xin, W_ih, b_ih, gi, cB, 3 * cDEC, cEMB + cENC);
  k_gemm_tr<8><<<dim3((cB / 128) * (3 * cDEC / 16)), dim3(32), 0, stream>>>(
      hidden, W_hh, b_hh, gh, cB, 3 * cDEC, cDEC);

  // 7) combine gates -> h_new (output tail + xcat[:,0:DEC])
  k_gru_combine<<<dim3(cB * cDEC / 256), dim3(256), 0, stream>>>(
      gi, gh, hidden, outh, xcat);
  k_build_xcat_rest<<<dim3(cB * (cENC + cEMB) / 256), dim3(256), 0, stream>>>(
      weighted, emb, xcat);

  // 8) pred = xcat @ fc_W^T + fc_b  (21 GFLOP / 164 MB weight stream, MT=16
  //    => exactly one pass over fc_W; 2000 blocks)
  k_gemm_tr<16><<<dim3((cB / 256) * (cV / 16)), dim3(32), 0, stream>>>(
      xcat, fc_W, fc_b, out, cB, cV, cDEC + cENC + cEMB);
}
